// Outlooker_89687507075724
// MI455X (gfx1250) — matrix operand order repaired and checked by reading, not compiled, not run on hardware
//
#include <hip/hip_runtime.h>
#include <hip/hip_bf16.h>
#include <math.h>

// ---------------- problem constants ----------------
#define BATCH   32
#define HH      28
#define WW      28
#define CC      192
#define NTOK    (BATCH * HH * WW)      // 25088 tokens (divisible by 64)
#define NHEADS  6
#define HDIM    32
#define KK2     9                      // 3x3 window
#define NA      486                    // HEADS * 9 * 9
#define NA_STRIDE 488                  // padded row stride (1952B = 122 x 16B)
#define NA_PAD  512                    // N padded to multiple of 64 (wave tile 16x64)
#define HID     576
#define KCHUNK  192                    // K staging chunk (192 | 576 both divisible)
#define ATT_SCALE 0.17677669529663687f // 32^-0.5
#define LN_EPS  1e-5f

typedef __attribute__((ext_vector_type(16))) _Float16 v16h;
typedef __attribute__((ext_vector_type(8)))  _Float16 v8h;
typedef __attribute__((ext_vector_type(8)))  float    v8f;
typedef __attribute__((ext_vector_type(4)))  unsigned int u32x4;
typedef __attribute__((ext_vector_type(8)))  unsigned int u32x8;

// ---------------------------------------------------------------------------
// Weight prep: fp32 W[K][N] row-major  ->  f16 Wt[Npad][K]  (N-major, padded)
// ---------------------------------------------------------------------------
__global__ void prep_weight_kernel(const float* __restrict__ W,
                                   _Float16* __restrict__ Wt,
                                   int K, int N, int Npad) {
    int idx = blockIdx.x * 256 + threadIdx.x;
    int total = Npad * K;
    if (idx >= total) return;
    int n = idx / K;
    int k = idx - n * K;
    Wt[idx] = (n < N) ? (_Float16)W[(size_t)k * N + n] : (_Float16)0.0f;
}

// ---------------------------------------------------------------------------
// LayerNorm over C=192, one wave32 per token, output f16
// ---------------------------------------------------------------------------
__global__ void layernorm_f16_kernel(const float* __restrict__ x,
                                     const float* __restrict__ g,
                                     const float* __restrict__ b,
                                     _Float16* __restrict__ out) {
    int tok  = blockIdx.x * blockDim.y + threadIdx.y;
    int lane = threadIdx.x;
    if (tok >= NTOK) return;
    const float* row = x + (size_t)tok * CC;

    float vals[6];
    float s = 0.0f;
#pragma unroll
    for (int i = 0; i < 6; ++i) { vals[i] = row[lane + 32 * i]; s += vals[i]; }
#pragma unroll
    for (int d = 16; d >= 1; d >>= 1) s += __shfl_xor(s, d, 32);
    float mean = s * (1.0f / (float)CC);

    float vs = 0.0f;
#pragma unroll
    for (int i = 0; i < 6; ++i) { float t = vals[i] - mean; vs += t * t; }
#pragma unroll
    for (int d = 16; d >= 1; d >>= 1) vs += __shfl_xor(vs, d, 32);
    float rstd = rsqrtf(vs * (1.0f / (float)CC) + LN_EPS);

    _Float16* orow = out + (size_t)tok * CC;
#pragma unroll
    for (int i = 0; i < 6; ++i) {
        int c = lane + 32 * i;
        orow[c] = (_Float16)((vals[i] - mean) * rstd * g[c] + b[c]);
    }
}

// ---------------------------------------------------------------------------
// WMMA GEMM: C[M x Nreal] (ldc stride) = A[M x K](f16,row) @ Bt[Npad x K](f16,N-major)
// Block = 4 waves; block tile = 64(M) x 64(N). Each wave: 16x64 (4 accums,
// shared A fragment). B slab (64 x KCHUNK f16 = 24KB) staged in LDS by the
// Tensor Data Mover: wave 0 issues one tensor_load_to_lds per K-chunk
// (2D tile: 192 halfs x 64 rows, row stride K), waits TENSORcnt, barrier.
// B fragments then come from LDS (ds_load_b128); MACs are
// v_wmma_f32_16x16x32_f16.
// EPI: 0 = store f16 (no bias)        1 = +bias, store f32
//      2 = +bias +residual(f32), f32  3 = +bias, exact GELU, store f16
// ---------------------------------------------------------------------------
template <int EPI>
__global__ void gemm_wmma_kernel(const _Float16* __restrict__ A,
                                 const _Float16* __restrict__ Bt,
                                 const float* __restrict__ bias,
                                 const float* __restrict__ res,
                                 float* __restrict__ outF,
                                 _Float16* __restrict__ outH,
                                 int M, int Nreal, int ldc, int K) {
    __shared__ __align__(64) _Float16 ldsB[64 * KCHUNK];   // 24 KB slab @ LDS base

    const int lane = threadIdx.x;                 // 0..31
    const int wave = threadIdx.y;                 // 0..3
    const int m0 = (blockIdx.x * 4 + wave) * 16;  // wave-uniform (M % 64 == 0)
    const int n0 = blockIdx.y * 64;

    const int half = lane >> 4;
    const int l15  = lane & 15;

    // A fragment rows: lane holds row m0+l15; e -> k=half*16+e
    const _Float16* arow = A + (size_t)(m0 + l15) * K + half * 16;

    v8f acc[4] = {{}, {}, {}, {}};

    for (int kc = 0; kc < K; kc += KCHUNK) {
        // ---- TDM stage: B rows n0..n0+63, K-range [kc, kc+KCHUNK) ----
        if (wave == 0) {
            unsigned long long ga =
                (unsigned long long)(Bt + (size_t)n0 * K + kc);
            u32x4 g0;
            g0[0] = 1u;                                   // count=1 (user D#)
            g0[1] = 0u;                                   // lds_addr = slab base
            g0[2] = (unsigned)ga;                         // global_addr[31:0]
            g0[3] = ((unsigned)(ga >> 32) & 0x01FFFFFFu)  // global_addr[56:32]
                    | (2u << 30);                         // type = 2 ("image")
            u32x8 g1;
            g1[0] = 1u << 16;                             // data_size=1 -> 2 bytes
            g1[1] = ((unsigned)KCHUNK & 0xFFFFu) << 16;   // tensor_dim0[15:0]
            g1[2] = ((unsigned)KCHUNK >> 16) | (64u << 16); // td0[31:16] | td1[15:0]
            g1[3] = ((unsigned)KCHUNK) << 16;             // td1[31:16]=0 | tile_dim0
            g1[4] = 64u;                                  // tile_dim1=64, tile_dim2=0
            g1[5] = (unsigned)K;                          // tensor_dim0_stride[31:0]
            g1[6] = 0u;
            g1[7] = 0u;
            asm volatile("tensor_load_to_lds %0, %1"
                         :: "s"(g0), "s"(g1) : "memory");
            asm volatile("s_wait_tensorcnt 0x0" ::: "memory");
        }
        __syncthreads();

        // ---- compute: 6 k-steps x 4 n-tiles = 24 WMMAs per chunk ----
#pragma unroll
        for (int kk = 0; kk < KCHUNK; kk += 32) {
            union { v16h v; v8h h[2]; } af;
            af.h[0] = *(const v8h*)(arow + kc + kk);
            af.h[1] = *(const v8h*)(arow + kc + kk + 8);
            // B fragment from LDS: lane -> col l15 of n-tile, k = half*16+0..15
            const _Float16* bbase = ldsB + (size_t)l15 * KCHUNK + half * 16 + kk;
#pragma unroll
            for (int nt = 0; nt < 4; ++nt) {
                v16h bf = *(const v16h*)(bbase + (size_t)nt * 16 * KCHUNK);
                acc[nt] = __builtin_amdgcn_wmma_f32_16x16x32_f16(
                              false, af.v, false, bf, (short)0, acc[nt], false, false);
            }
        }
        __syncthreads();   // slab reuse fence
    }

    // ---- epilogue (after all WMMAs; divergence now safe) ----
#pragma unroll
    for (int nt = 0; nt < 4; ++nt) {
        const int n = n0 + nt * 16 + l15;
        if (n >= Nreal) continue;
        const float bv = (EPI == 0) ? 0.0f : bias[n];
#pragma unroll
        for (int v = 0; v < 8; ++v) {
            int m = m0 + v + 8 * half;            // C/D layout: M = vgpr + 8*half
            float val = acc[nt][v] + bv;
            size_t idx = (size_t)m * ldc + n;
            if (EPI == 2) val += res[idx];
            if (EPI == 3) val = 0.5f * val * (1.0f + erff(val * 0.70710678118654752f));
            if (EPI == 0 || EPI == 3) outH[idx] = (_Float16)val;
            else                      outF[idx] = val;
        }
    }
}

// ---------------------------------------------------------------------------
// Softmax over groups of 9 (attn layout: [NTOK][NA_STRIDE], groups of 9 in
// the leading 486 of each row)
// ---------------------------------------------------------------------------
__global__ void softmax9_kernel(float* __restrict__ a) {
    int idx = blockIdx.x * 256 + threadIdx.x;     // global group index
    const int ngroups = NTOK * NHEADS * KK2;      // 54 groups per token
    if (idx >= ngroups) return;
    int tok = idx / 54;
    int g   = idx - tok * 54;
    float* p = a + (size_t)tok * NA_STRIDE + g * 9;

    float v[9], m = -3.4e38f;
#pragma unroll
    for (int j = 0; j < 9; ++j) { v[j] = p[j] * ATT_SCALE; m = fmaxf(m, v[j]); }
    float s = 0.0f;
#pragma unroll
    for (int j = 0; j < 9; ++j) { v[j] = __expf(v[j] - m); s += v[j]; }
    float r = 1.0f / s;
#pragma unroll
    for (int j = 0; j < 9; ++j) p[j] = v[j] * r;
}

// ---------------------------------------------------------------------------
// Outlook attention + fold, gather form (no atomics):
//   y[b,Y,X,c] = sum_k sum_j a[b, src(k), head(c), k, j] * v[b, src(k)+off(j), c]
// One block (192 threads = 6 waves) per output pixel. Per-block LDS staging
// via CDNA5 async copies (global_load_async_to_lds_b128 + ASYNCcnt):
//   - 9 candidate a-rows (488 floats each, 122 x 16B chunks)
//   - 5x5 v-row neighborhood (192 halfs each, 24 x 16B chunks)
// Compute reads only LDS; output stored f16.
// ---------------------------------------------------------------------------
#define SA_BYTES (9 * NA_STRIDE * 4)              // 17568
#define SV_BYTES (25 * CC * 2)                    // 9600
__global__ void outlook_av_fold_kernel(const float* __restrict__ a,
                                       const _Float16* __restrict__ v,
                                       _Float16* __restrict__ y) {
    __shared__ __align__(16) unsigned char smem[SA_BYTES + SV_BYTES];

    const int c   = threadIdx.x;                  // 0..191
    const int pix = blockIdx.x;                   // 0..NTOK-1
    const int b   = pix / (HH * WW);
    const int rem = pix - b * (HH * WW);
    const int Y = rem / WW, X = rem - (rem / WW) * WW;
    const int head = c >> 5;

    // ---- async stage: a-rows for the 9 source-pixel slots ----
    for (int t = c; t < 9 * 122; t += CC) {
        const int slot = t / 122, ch = t - slot * 122;
        const int ki = slot / 3, kj = slot - ki * 3;
        const int sy = Y + 1 - ki, sx = X + 1 - kj;
        if ((unsigned)sy < (unsigned)HH && (unsigned)sx < (unsigned)WW) {
            const int stok = (b * HH + sy) * WW + sx;
            unsigned lds_off = (unsigned)(slot * (NA_STRIDE * 4) + ch * 16);
            unsigned long long ga = (unsigned long long)
                ((const char*)(a + (size_t)stok * NA_STRIDE) + ch * 16);
            asm volatile("global_load_async_to_lds_b128 %0, %1, off"
                         :: "v"(lds_off), "v"(ga) : "memory");
        }
    }
    // ---- async stage: v-rows for the 5x5 neighborhood ----
    for (int t = c; t < 25 * 24; t += CC) {
        const int slot = t / 24, ch = t - slot * 24;
        const int vy = Y + slot / 5 - 2, vx = X + slot % 5 - 2;
        if ((unsigned)vy < (unsigned)HH && (unsigned)vx < (unsigned)WW) {
            const size_t vtok = (size_t)(b * HH + vy) * WW + vx;
            unsigned lds_off = (unsigned)(SA_BYTES + slot * (CC * 2) + ch * 16);
            unsigned long long ga = (unsigned long long)
                ((const char*)(v + vtok * CC) + ch * 16);
            asm volatile("global_load_async_to_lds_b128 %0, %1, off"
                         :: "v"(lds_off), "v"(ga) : "memory");
        }
    }
    asm volatile("s_wait_asynccnt 0x0" ::: "memory");
    __syncthreads();

    const float*    sa = (const float*)smem;
    const _Float16* sv = (const _Float16*)(smem + SA_BYTES);

    float acc = 0.0f;
#pragma unroll
    for (int k = 0; k < 9; ++k) {
        const int ki = k / 3, kj = k % 3;
        const int sy = Y + 1 - ki, sx = X + 1 - kj;
        if ((unsigned)sy < (unsigned)HH && (unsigned)sx < (unsigned)WW) {
            const float* arow = sa + k * NA_STRIDE + head * 81 + k * 9;
#pragma unroll
            for (int j = 0; j < 9; ++j) {
                const int jy = j / 3, jx = j % 3;
                const int vy = sy + jy - 1, vx = sx + jx - 1;
                if ((unsigned)vy < (unsigned)HH && (unsigned)vx < (unsigned)WW) {
                    const int slot = (jy - ki + 2) * 5 + (jx - kj + 2);
                    acc += arow[j] * (float)sv[slot * CC + c];
                }
            }
        }
    }
    y[(size_t)pix * CC + c] = (_Float16)acc;
}

// ---------------------------------------------------------------------------
// Host-side launch
// ---------------------------------------------------------------------------
extern "C" void kernel_launch(void* const* d_in, const int* in_sizes, int n_in,
                              void* d_out, int out_size, void* d_ws, size_t ws_size,
                              hipStream_t stream) {
    const float* x     = (const float*)d_in[0];
    const float* ln1_g = (const float*)d_in[1];
    const float* ln1_b = (const float*)d_in[2];
    const float* Wv    = (const float*)d_in[3];
    const float* Wa    = (const float*)d_in[4];
    const float* ba    = (const float*)d_in[5];
    const float* Wp    = (const float*)d_in[6];
    const float* bp    = (const float*)d_in[7];
    const float* ln2_g = (const float*)d_in[8];
    const float* ln2_b = (const float*)d_in[9];
    const float* W1    = (const float*)d_in[10];
    const float* b1    = (const float*)d_in[11];
    const float* W2    = (const float*)d_in[12];
    const float* b2    = (const float*)d_in[13];
    float* out = (float*)d_out;

    // ---- workspace carve-out (256B aligned) ----
    char* ws = (char*)d_ws;
    size_t off = 0;
    auto take = [&](size_t bytes) -> char* {
        char* p = ws + off;
        off = (off + bytes + 255) & ~(size_t)255;
        return p;
    };
    _Float16* ln1_h = (_Float16*)take((size_t)NTOK * CC  * 2);
    _Float16* v_h   = (_Float16*)take((size_t)NTOK * CC  * 2);
    float*    attn  = (float*)   take((size_t)NTOK * NA_STRIDE * 4);
    _Float16* y_h   = (_Float16*)take((size_t)NTOK * CC  * 2);
    float*    xmid  = (float*)   take((size_t)NTOK * CC  * 4);
    _Float16* ln2_h = (_Float16*)take((size_t)NTOK * CC  * 2);
    _Float16* hid_h = (_Float16*)take((size_t)NTOK * HID * 2);
    _Float16* wv_t  = (_Float16*)take((size_t)CC     * CC * 2);
    _Float16* wa_t  = (_Float16*)take((size_t)NA_PAD * CC * 2);
    _Float16* wp_t  = (_Float16*)take((size_t)CC     * CC * 2);
    _Float16* w1_t  = (_Float16*)take((size_t)HID    * CC * 2);
    _Float16* w2_t  = (_Float16*)take((size_t)CC     * HID * 2);

    // ---- 0) weights -> f16, N-major ----
    auto prep = [&](const float* W, _Float16* Wt, int K, int N, int Npad) {
        int total = Npad * K;
        prep_weight_kernel<<<(total + 255) / 256, 256, 0, stream>>>(W, Wt, K, N, Npad);
    };
    prep(Wv, wv_t, CC,  CC,  CC);
    prep(Wa, wa_t, CC,  NA,  NA_PAD);
    prep(Wp, wp_t, CC,  CC,  CC);
    prep(W1, w1_t, CC,  HID, HID);
    prep(W2, w2_t, HID, CC,  CC);

    const dim3 lnBlk(32, 8);
    const int  lnGrid = (NTOK + 7) / 8;
    const dim3 gBlk(32, 4);
    const int  gmx = NTOK / 64;                   // 392 M-blocks

    // ---- 1) LN1 -> f16 ----
    layernorm_f16_kernel<<<lnGrid, lnBlk, 0, stream>>>(x, ln1_g, ln1_b, ln1_h);

    // ---- 2) v = ln1 @ Wv  (f16 out) ----
    gemm_wmma_kernel<0><<<dim3(gmx, CC / 64), gBlk, 0, stream>>>(
        ln1_h, wv_t, nullptr, nullptr, nullptr, v_h, NTOK, CC, CC, CC);

    // ---- 3) attn logits = ln1 @ Wa + ba  (f32, N=486, ldc=488, pad N=512) ----
    gemm_wmma_kernel<1><<<dim3(gmx, NA_PAD / 64), gBlk, 0, stream>>>(
        ln1_h, wa_t, ba, nullptr, attn, nullptr, NTOK, NA, NA_STRIDE, CC);

    // ---- 4) softmax over last-9 ----
    {
        int ngroups = NTOK * NHEADS * KK2;
        softmax9_kernel<<<(ngroups + 255) / 256, 256, 0, stream>>>(attn);
    }

    // ---- 5) attention apply + fold (gather) -> f16 ----
    outlook_av_fold_kernel<<<NTOK, CC, 0, stream>>>(attn, v_h, y_h);

    // ---- 6) xmid = x + y @ Wp + bp ----
    gemm_wmma_kernel<2><<<dim3(gmx, CC / 64), gBlk, 0, stream>>>(
        y_h, wp_t, bp, x, xmid, nullptr, NTOK, CC, CC, CC);

    // ---- 7) LN2 -> f16 ----
    layernorm_f16_kernel<<<lnGrid, lnBlk, 0, stream>>>(xmid, ln2_g, ln2_b, ln2_h);

    // ---- 8) hid = gelu(ln2 @ W1 + b1) -> f16 ----
    gemm_wmma_kernel<3><<<dim3(gmx, HID / 64), gBlk, 0, stream>>>(
        ln2_h, w1_t, b1, nullptr, nullptr, hid_h, NTOK, HID, HID, CC);

    // ---- 9) out = xmid + hid @ W2 + b2 ----
    gemm_wmma_kernel<2><<<dim3(gmx, CC / 64), gBlk, 0, stream>>>(
        hid_h, w2_t, b2, xmid, out, nullptr, NTOK, CC, CC, HID);
}
